// MQ2_76974403879739
// MI455X (gfx1250) — compile-verified, matrix operand order checked by reading, not run on hardware
//
#include <hip/hip_runtime.h>
#include <hip/hip_bf16.h>
#include <math.h>

// ---------------------------------------------------------------------------
// 2-layer LSTM (B=64, T=256, D=H=1024) + gather-last + FC/BN/ReLU + logits
// Strategy: bf16 WMMA (v_wmma_f32_16x16x32_bf16) for the recurrent GEMMs.
// Weights are converted ONCE into a fragment-native swizzled layout so the
// per-timestep GEMM is pure coalesced-load -> wmma. The state-update kernel
// writes h directly back in swizzled A-fragment form for the next step.
// Both layers' bf16 weights (32MB) are L2-resident (192MB L2) -> per-step
// GEMM streams from L2, never HBM.
// ---------------------------------------------------------------------------

typedef __bf16 bf16_t;
typedef bf16_t v16bf __attribute__((ext_vector_type(16)));
typedef float  v8f   __attribute__((ext_vector_type(8)));

#define BB   64
#define TT   256
#define DD   1024
#define HH   1024
#define FOURH 4096
#define KTOT 2048            // D + H (layer0) == H + H (layer1)
#define FCN  512
#define OUTN 11
#define BN_EPS 1e-5f

// A-fragment element mapping (16-bit A, 16x32, per ISA 7.12.2):
//   M = lane & 15
//   K(e,lane) = (e>=8 ? 16:0) + (lane>=16 ? 8:0) + 2*((e>>1)&3) + (e&1)
__device__ __forceinline__ int frag_k(int e, int lane) {
    int v = e >> 1;
    return ((v >= 4) ? 16 : 0) + ((lane >= 16) ? 8 : 0) + ((v & 3) << 1) + (e & 1);
}

__device__ __forceinline__ float sigmf(float x) { return 1.0f / (1.0f + __expf(-x)); }

// ---------------------------------------------------------------------------
// Swizzle W [2048][4096] f32 row-major -> bf16 fragments: dst[nt][kt][lane][e]
// where element = W[kt*32 + K(e,lane)][nt*16 + (lane&15)]   (B-operand layout)
// ---------------------------------------------------------------------------
__global__ void swz_weights(const float* __restrict__ src, bf16_t* __restrict__ dst) {
    int idx  = blockIdx.x * blockDim.x + threadIdx.x;      // < 2048*4096
    int e    = idx & 15;
    int lane = (idx >> 4) & 31;
    int kt   = (idx >> 9) & 63;
    int nt   = idx >> 15;                                  // 0..255
    int K    = frag_k(e, lane);
    int row  = kt * 32 + K;
    int col  = nt * 16 + (lane & 15);
    dst[idx] = (bf16_t)src[(size_t)row * FOURH + col];
}

// ---------------------------------------------------------------------------
// Swizzle x [B][T][D] f32 -> x_sw[t][mt][kt][lane][e]  (A-operand layout)
//   batch b = mt*16 + (lane&15), feature d = kt*32 + K(e,lane)
// ---------------------------------------------------------------------------
__global__ void swz_x(const float* __restrict__ src, bf16_t* __restrict__ dst) {
    int idx  = blockIdx.x * blockDim.x + threadIdx.x;      // < T*4*32*512
    int e    = idx & 15;
    int lane = (idx >> 4) & 31;
    int kt   = (idx >> 9) & 31;
    int mt   = (idx >> 14) & 3;
    int t    = idx >> 16;
    int b    = mt * 16 + (lane & 15);
    int d    = kt * 32 + frag_k(e, lane);
    dst[idx] = (bf16_t)src[((size_t)b * TT + t) * DD + d];
}

// ---------------------------------------------------------------------------
// Zero c-state, swizzled h-state, and the gathered-h buffer.
// ---------------------------------------------------------------------------
__global__ void init_state(float* __restrict__ c, bf16_t* __restrict__ h_sw,
                           float* __restrict__ last_h) {
    int idx = blockIdx.x * blockDim.x + threadIdx.x;       // < 65536
    c[idx]      = 0.0f;
    h_sw[idx]   = (bf16_t)0.0f;
    last_h[idx] = 0.0f;
}

// ---------------------------------------------------------------------------
// One recurrence step GEMM: z = [x_t , h_{t-1}] @ W + b      [64 x 4096]
// Grid: 128 blocks x 256 thr = 1024 waves; wave -> one 16x16 tile (mt,nt).
// K = 2048 = 32 ktiles from xin + 32 ktiles from hin; 64 WMMAs per wave.
// All operands pre-swizzled: each lane loads contiguous 32B per fragment.
// ---------------------------------------------------------------------------
__global__ void lstm_gemm_step(const bf16_t* __restrict__ xin,   // [4][32][512]
                               const bf16_t* __restrict__ hin,   // [4][32][512]
                               const bf16_t* __restrict__ Wsw,   // [256][64][512]
                               const float*  __restrict__ bias,  // [4096]
                               float* __restrict__ z)            // [64][4096]
{
    const int lane = threadIdx.x & 31;
    const int wave = threadIdx.x >> 5;
    const int tile = blockIdx.x * (blockDim.x >> 5) + wave;    // 0..1023
    const int mt   = tile >> 8;                                // 0..3   (batch tile)
    const int nt   = tile & 255;                               // 0..255 (gate-col tile)

    const bf16_t* wp = Wsw + (size_t)nt * (64 * 512) + lane * 16;
    const bf16_t* xp = xin + (size_t)mt * (32 * 512) + lane * 16;
    const bf16_t* hp = hin + (size_t)mt * (32 * 512) + lane * 16;

    v8f acc = {0.f, 0.f, 0.f, 0.f, 0.f, 0.f, 0.f, 0.f};

#pragma unroll 4
    for (int kt = 0; kt < 32; ++kt) {
        v16bf a = *(const v16bf*)(xp + kt * 512);
        v16bf b = *(const v16bf*)(wp + kt * 512);
        __builtin_prefetch((const void*)(wp + (kt + 8) * 512), 0, 1);
        acc = __builtin_amdgcn_wmma_f32_16x16x32_bf16(false, a, false, b,
                                                      (short)0, acc, false, false);
    }
#pragma unroll 4
    for (int kt = 0; kt < 32; ++kt) {
        v16bf a = *(const v16bf*)(hp + kt * 512);
        v16bf b = *(const v16bf*)(wp + (32 + kt) * 512);
        acc = __builtin_amdgcn_wmma_f32_16x16x32_bf16(false, a, false, b,
                                                      (short)0, acc, false, false);
    }

    // C/D layout: N = lane&15, M = r + (lane>=16 ? 8 : 0)
    const int n     = nt * 16 + (lane & 15);
    const float bv  = bias[n];
    const int  row0 = mt * 16 + ((lane < 16) ? 0 : 8);
#pragma unroll
    for (int r = 0; r < 8; ++r) {
        z[(size_t)(row0 + r) * FOURH + n] = acc[r] + bv;
    }
}

// ---------------------------------------------------------------------------
// Gate nonlinearities + state update. Writes new h in swizzled A-fragment
// form (ready for next step), optionally records h1[t] (layer0) or scatters
// h2 at t==seq_end[b]-1 (layer1).
// ---------------------------------------------------------------------------
__global__ void lstm_update_step(const float* __restrict__ z,       // [64][4096]
                                 float* __restrict__ c,             // [64][1024]
                                 bf16_t* __restrict__ h_sw,         // [4][32][512]
                                 bf16_t* __restrict__ h_hist_sw,    // or null
                                 const int* __restrict__ seq_end,   // or null
                                 float* __restrict__ last_h,        // or null
                                 int t)
{
    int idx = blockIdx.x * blockDim.x + threadIdx.x;   // < 65536
    int b = idx >> 10;
    int u = idx & 1023;

    const float* zb = z + (size_t)b * FOURH;
    float zi = zb[u];
    float zj = zb[u + 1024];
    float zf = zb[u + 2048];
    float zo = zb[u + 3072];

    float cn = c[idx] * sigmf(zf + 1.0f) + sigmf(zi) * tanhf(zj);
    float hn = tanhf(cn) * sigmf(zo);
    c[idx] = cn;

    // inverse fragment mapping: (b,u) -> (mt, kt, lane, e)
    int mt = b >> 4, M = b & 15;
    int kt = u >> 5, K = u & 31;
    int k16  = K & 15;
    int lane = M + ((k16 >= 8) ? 16 : 0);
    int k8   = k16 & 7;
    int v    = ((K >= 16) ? 4 : 0) + (k8 >> 1);
    int e    = (v << 1) + (k8 & 1);
    int sidx = ((mt * 32 + kt) * 32 + lane) * 16 + e;

    bf16_t hb = (bf16_t)hn;
    h_sw[sidx] = hb;
    if (h_hist_sw) h_hist_sw[sidx] = hb;
    if (seq_end && t == (seq_end[b] - 1)) last_h[idx] = hn;
}

// ---------------------------------------------------------------------------
// FC1 (1024->512) + BN + ReLU; tiny -> plain f32.
// ---------------------------------------------------------------------------
__global__ void fc1_bn_relu(const float* __restrict__ last_h,  // [64][1024]
                            const float* __restrict__ W,       // [1024][512]
                            const float* __restrict__ bias,
                            const float* __restrict__ gamma,
                            const float* __restrict__ beta,
                            const float* __restrict__ mean,
                            const float* __restrict__ var,
                            float* __restrict__ out)           // [64][512]
{
    int idx = blockIdx.x * blockDim.x + threadIdx.x;   // < 64*512
    int b = idx >> 9;
    int j = idx & 511;
    const float* h = last_h + (size_t)b * HH;
    float acc = bias[j];
    for (int k = 0; k < HH; ++k) acc = fmaf(h[k], W[(size_t)k * FCN + j], acc);
    acc = (acc - mean[j]) * rsqrtf(var[j] + BN_EPS) * gamma[j] + beta[j];
    out[idx] = fmaxf(acc, 0.0f);
}

__global__ void logits_kernel(const float* __restrict__ fc1,   // [64][512]
                              const float* __restrict__ W,     // [512][11]
                              const float* __restrict__ bias,
                              float* __restrict__ out)         // [64][11]
{
    int idx = blockIdx.x * blockDim.x + threadIdx.x;
    if (idx >= BB * OUTN) return;
    int b = idx / OUTN, o = idx % OUTN;
    const float* f = fc1 + (size_t)b * FCN;
    float acc = bias[o];
    for (int k = 0; k < FCN; ++k) acc = fmaf(f[k], W[(size_t)k * OUTN + o], acc);
    out[idx] = acc;
}

// ---------------------------------------------------------------------------
extern "C" void kernel_launch(void* const* d_in, const int* in_sizes, int n_in,
                              void* d_out, int out_size, void* d_ws, size_t ws_size,
                              hipStream_t stream) {
    const float* x       = (const float*)d_in[0];
    const int*   seq_end = (const int*)  d_in[1];
    const float* W0      = (const float*)d_in[2];
    const float* b0      = (const float*)d_in[3];
    const float* W1      = (const float*)d_in[4];
    const float* b1      = (const float*)d_in[5];
    const float* fc1_W   = (const float*)d_in[6];
    const float* fc1_b   = (const float*)d_in[7];
    const float* gamma   = (const float*)d_in[8];
    const float* beta    = (const float*)d_in[9];
    const float* mean    = (const float*)d_in[10];
    const float* var     = (const float*)d_in[11];
    const float* lW      = (const float*)d_in[12];
    const float* lb      = (const float*)d_in[13];
    float*       out     = (float*)d_out;

    // ---- workspace carve-out (~98 MB) ----
    char*  ws  = (char*)d_ws;
    size_t off = 0;
    auto carve = [&](size_t bytes) -> void* {
        void* p = ws + off;
        off += (bytes + 255) & ~(size_t)255;
        return p;
    };
    const size_t WELEMS = (size_t)KTOT * FOURH;          // 8M elems / matrix
    const size_t FRAG_T = (size_t)4 * 32 * 512;          // 65536 elems / timestep
    bf16_t* W0sw   = (bf16_t*)carve(WELEMS * sizeof(bf16_t));        // 16 MB
    bf16_t* W1sw   = (bf16_t*)carve(WELEMS * sizeof(bf16_t));        // 16 MB
    bf16_t* x_sw   = (bf16_t*)carve((size_t)TT * FRAG_T * 2);        // 32 MB
    bf16_t* h1_sw  = (bf16_t*)carve((size_t)TT * FRAG_T * 2);        // 32 MB
    bf16_t* h_sw   = (bf16_t*)carve(FRAG_T * 2);                     // 128 KB
    float*  cstate = (float*) carve((size_t)BB * HH * 4);            // 256 KB
    float*  zbuf   = (float*) carve((size_t)BB * FOURH * 4);         // 1 MB
    float*  last_h = (float*) carve((size_t)BB * HH * 4);            // 256 KB
    float*  fc1v   = (float*) carve((size_t)BB * FCN * 4);           // 128 KB
    (void)ws_size; (void)in_sizes; (void)n_in; (void)out_size;

    // ---- one-time data layout transforms ----
    swz_weights<<<WELEMS / 256, 256, 0, stream>>>(W0, W0sw);
    swz_weights<<<WELEMS / 256, 256, 0, stream>>>(W1, W1sw);
    swz_x<<<(size_t)TT * FRAG_T / 256, 256, 0, stream>>>(x, x_sw);

    // ---- layer 0 ----
    init_state<<<256, 256, 0, stream>>>(cstate, h_sw, last_h);
    for (int t = 0; t < TT; ++t) {
        lstm_gemm_step<<<128, 256, 0, stream>>>(x_sw + (size_t)t * FRAG_T, h_sw,
                                                W0sw, b0, zbuf);
        lstm_update_step<<<256, 256, 0, stream>>>(zbuf, cstate, h_sw,
                                                  h1_sw + (size_t)t * FRAG_T,
                                                  nullptr, nullptr, t);
    }

    // ---- layer 1 (+ gather h2 at seq_end-1) ----
    init_state<<<256, 256, 0, stream>>>(cstate, h_sw, last_h);
    for (int t = 0; t < TT; ++t) {
        lstm_gemm_step<<<128, 256, 0, stream>>>(h1_sw + (size_t)t * FRAG_T, h_sw,
                                                W1sw, b1, zbuf);
        lstm_update_step<<<256, 256, 0, stream>>>(zbuf, cstate, h_sw,
                                                  nullptr, seq_end, last_h, t);
    }

    // ---- tail ----
    fc1_bn_relu<<<(BB * FCN) / 256, 256, 0, stream>>>(last_h, fc1_W, fc1_b,
                                                      gamma, beta, mean, var, fc1v);
    logits_kernel<<<(BB * OUTN + 255) / 256, 256, 0, stream>>>(fc1v, lW, lb, out);
}
